// CromLinear_3521873183500
// MI455X (gfx1250) — matrix operand order repaired and checked by reading, not compiled, not run on hardware
//
#include <hip/hip_runtime.h>

// ---------------------------------------------------------------------------
// CROM quantized linear for MI455X (gfx1250, wave32, WMMA + TDM).
//   out[M,N] = x[M,K] @ W[K,N] + bias ;  W[k][n] = codebook[indices[k*64+n/64]][n%64]
// M=1024, N=4096, K=4096.  ~1000 FLOP/byte -> compute-bound -> bf16 WMMA f32 acc.
// Tile staging via Tensor Data Mover (tensor_load_to_lds, TENSORcnt),
// double-buffered LDS, padded rows emitted by TDM's pad fields.
// ---------------------------------------------------------------------------

typedef __attribute__((ext_vector_type(16))) __bf16   v16bf;
typedef __attribute__((ext_vector_type(8)))  float    v8f;
typedef __attribute__((ext_vector_type(4)))  uint32_t v4u;
typedef __attribute__((ext_vector_type(8)))  uint32_t v8u;

#define M_DIM 1024
#define N_DIM 4096
#define K_DIM 4096

#define BM 128
#define BN 128
#define BK 64
#define LDSS  72            // halves per LDS row: 64 data + 8 pad (144B = 36 banks; 16B aligned)
#define AT    (BM * LDSS)   // halves per A tile (18 KB)
#define BT    (BN * LDSS)   // halves per B tile (18 KB)
#define SBUF  (AT + BT)     // halves per double-buffer slot
#define NITER (K_DIM / BK)  // 64

// ---------------------------------------------------------------------------
// Kernel 1: decode weight into Wt[n][k] (N-major, K contiguous) as bf16.
//   - index loads uniform across 64 consecutive n (wave broadcast)
//   - codebook gathers coalesced; one 16B store per thread
// ---------------------------------------------------------------------------
__global__ __launch_bounds__(256) void crom_dequant(
    const float* __restrict__ codebook,   // [256][64]
    const int*   __restrict__ indices,    // [262144]
    __bf16*      __restrict__ wt)         // [N][K]
{
    const int n  = blockIdx.x * 256 + threadIdx.x;
    const int k0 = blockIdx.y * 8;
    const int j  = n >> 6;
    const int h  = n & 63;

    union { __bf16 h8[8]; uint4 u; } pack;
#pragma unroll
    for (int kk = 0; kk < 8; ++kk) {
        const int idx = indices[(size_t)(k0 + kk) * 64 + j];
        pack.h8[kk] = (__bf16)codebook[idx * 64 + h];
    }
    *(uint4*)(&wt[(size_t)n * K_DIM + k0]) = pack.u;
}

// ---------------------------------------------------------------------------
// Kernel 2: x f32 -> bf16 once (GEMM re-reads x 32x from L2; shrink it first)
// ---------------------------------------------------------------------------
__global__ __launch_bounds__(256) void crom_xconv(
    const float* __restrict__ x, __bf16* __restrict__ xb)
{
    const size_t i = ((size_t)blockIdx.x * 256 + threadIdx.x) * 8;
    const float4 a = *(const float4*)(x + i);
    const float4 b = *(const float4*)(x + i + 4);
    union { __bf16 h8[8]; uint4 u; } p;
    p.h8[0] = (__bf16)a.x; p.h8[1] = (__bf16)a.y;
    p.h8[2] = (__bf16)a.z; p.h8[3] = (__bf16)a.w;
    p.h8[4] = (__bf16)b.x; p.h8[5] = (__bf16)b.y;
    p.h8[6] = (__bf16)b.z; p.h8[7] = (__bf16)b.w;
    *(uint4*)(xb + i) = p.u;
}

// ---------------------------------------------------------------------------
// TDM helpers: build D# (ISA 8.3/8.4) for a 2D tile [128 rows x 64 halves],
// row stride 4096 halves, dest LDS rows padded 128B data + 16B pad.
// ---------------------------------------------------------------------------
__device__ __forceinline__ uint32_t lds_byte_off(const void* p) {
    // generic LDS pointer = {aperture_hi32, lds_offset_lo32}
    return (uint32_t)(uintptr_t)p;
}

__device__ __forceinline__ void tdm_load_tile(uint32_t lds_addr, const void* gptr) {
    const uint64_t ga = (uint64_t)(uintptr_t)gptr;
    v4u g0;
    g0[0] = 1u;                                   // count=1, is_restore=0, gather=0
    g0[1] = lds_addr;                             // LDS byte address
    g0[2] = (uint32_t)ga;                         // global_addr[31:0]
    g0[3] = (uint32_t)((ga >> 32) & 0x01FFFFFFu)  // global_addr[56:32]
            | (2u << 30);                         // type=2 ("image")
    v8u g1;
    g1[0] = (1u << 16)      // data_size = 2 bytes
          | (1u << 20)      // pad_enable
          | (4u << 22)      // pad_interval: 32 DWORDs (=128B = 64 halves)
          | (3u << 25);     // pad_amount:   4 DWORDs (=16B  =  8 halves)
    g1[1] = (uint32_t)(K_DIM & 0xFFFF) << 16;     // tensor_dim0 lo16 (4096)
    g1[2] = 0;                                    // tensor_dim0 hi16=0, tensor_dim1 lo16=0
    g1[3] = 0x10u | ((uint32_t)BK << 16);         // tensor_dim1 hi16 (1<<20 rows), tile_dim0=64
    g1[4] = (uint32_t)BM;                         // tile_dim1=128, tile_dim2=0 (2D)
    g1[5] = (uint32_t)K_DIM;                      // tensor_dim0_stride lo32 = 4096
    g1[6] = 0;
    g1[7] = 0;
    const v4u gz = {0u, 0u, 0u, 0u};              // groups 2/3: unused (tile_dim3/4 = 0)
    asm volatile("tensor_load_to_lds %0, %1, %2, %3"
                 :: "s"(g0), "s"(g1), "s"(gz), "s"(gz)
                 : "memory");
}

template <int N>
__device__ __forceinline__ void wait_tensorcnt() {
#if __has_builtin(__builtin_amdgcn_s_wait_tensorcnt)
    __builtin_amdgcn_s_wait_tensorcnt(N);
#else
    asm volatile("s_wait_tensorcnt %0" :: "i"(N) : "memory");
#endif
}

// ---------------------------------------------------------------------------
// Kernel 3: bf16 WMMA GEMM, TDM-staged, double-buffered.
// 256 threads = 8 waves (2 along M x 4 along N); wave tile 64x32 = 4x2 WMMA tiles.
// Per staged K-slab of 64: 2 half-steps x 8 WMMAs.
// A frag (16x32 bf16): lane l holds row M=l%16; K chunks {g*8, g*8+16..} (g=l/16).
// B frag (32x16):      lane l holds col N=l%16; K chunks {g*8, g*8+16..} (g=l/16).
// ---------------------------------------------------------------------------
__global__ __launch_bounds__(256) void crom_gemm(
    const __bf16* __restrict__ xb,    // [M][K] bf16
    const __bf16* __restrict__ wt,    // [N][K] bf16
    const float*  __restrict__ bias,  // [N]
    float*        __restrict__ out)   // [M][N] f32
{
    __shared__ __bf16 lds[2 * SBUF];  // 72 KB of the 320 KB WGP pool

    const int t    = threadIdx.x;
    const int lane = t & 31;
    const int wave = t >> 5;
    const int wm   = wave & 1;
    const int wn   = wave >> 1;
    const int m0   = blockIdx.y * BM;
    const int n0   = blockIdx.x * BN;
    const int g    = lane >> 4;
    const int lm   = lane & 15;

    const char* aG = (const char*)xb + (size_t)m0 * K_DIM * 2;   // + k*2 per slab
    const char* bG = (const char*)wt + (size_t)n0 * K_DIM * 2;

    v8f acc[4][2];
#pragma unroll
    for (int mi = 0; mi < 4; ++mi)
#pragma unroll
        for (int ni = 0; ni < 2; ++ni)
            acc[mi][ni] = (v8f){0.f, 0.f, 0.f, 0.f, 0.f, 0.f, 0.f, 0.f};

    // Prologue: stage slab 0 into buffer 0 (wave 0 -> A tile, wave 1 -> B tile)
    if (wave == 0)      tdm_load_tile(lds_byte_off(&lds[0]),  aG);
    else if (wave == 1) tdm_load_tile(lds_byte_off(&lds[AT]), bG);

    for (int it = 0; it < NITER; ++it) {
        const int cur = it & 1;

        // Stage next slab into the other buffer; keep exactly one TDM in flight.
        if (it + 1 < NITER) {
            const size_t koff = (size_t)(it + 1) * BK * 2;
            const int nxt = cur ^ 1;
            if (wave == 0)      tdm_load_tile(lds_byte_off(&lds[nxt * SBUF]),      aG + koff);
            else if (wave == 1) tdm_load_tile(lds_byte_off(&lds[nxt * SBUF + AT]), bG + koff);
            if (wave < 2) wait_tensorcnt<1>();
        } else if (wave < 2) {
            wait_tensorcnt<0>();
        }
        __syncthreads();   // publish buffer `cur` to all waves

        const __bf16* Ab = &lds[cur * SBUF];
        const __bf16* Bb = Ab + AT;

#pragma unroll
        for (int kh = 0; kh < 2; ++kh) {
            v16bf af[4];
#pragma unroll
            for (int mi = 0; mi < 4; ++mi) {
                const __bf16* p = &Ab[(wm * 64 + mi * 16 + lm) * LDSS + kh * 32 + g * 8];
                union { uint4 u[2]; v16bf v; } fa;
                fa.u[0] = *(const uint4*)(p);        // K: g*8   .. g*8+7
                fa.u[1] = *(const uint4*)(p + 16);   // K: g*8+16.. g*8+23
                af[mi] = fa.v;
            }
            v16bf bfr[2];
#pragma unroll
            for (int ni = 0; ni < 2; ++ni) {
                const __bf16* p = &Bb[(wn * 32 + ni * 16 + lm) * LDSS + kh * 32 + g * 8];
                union { uint4 u[2]; v16bf v; } fb;
                fb.u[0] = *(const uint4*)(p);        // K: g*8   .. g*8+7
                fb.u[1] = *(const uint4*)(p + 16);   // K: g*8+16.. g*8+23
                bfr[ni] = fb.v;
            }
#pragma unroll
            for (int mi = 0; mi < 4; ++mi)
#pragma unroll
                for (int ni = 0; ni < 2; ++ni)
                    acc[mi][ni] = __builtin_amdgcn_wmma_f32_16x16x32_bf16(
                        false, af[mi], false, bfr[ni],
                        (short)0, acc[mi][ni], false, false);
        }
        __syncthreads();   // all waves done with `cur` before it is restaged
    }

    // Epilogue: D layout -> lane covers N=lm, M = g*8 + v; add bias, f32 out
#pragma unroll
    for (int ni = 0; ni < 2; ++ni) {
        const int n   = n0 + wn * 32 + ni * 16 + lm;
        const float bv = bias[n];
#pragma unroll
        for (int mi = 0; mi < 4; ++mi) {
            const int mb = m0 + wm * 64 + mi * 16 + g * 8;
#pragma unroll
            for (int v = 0; v < 8; ++v)
                out[(size_t)(mb + v) * N_DIM + n] = acc[mi][ni][v] + bv;
        }
    }
}

// ---------------------------------------------------------------------------
// Launch. Inputs: x[0], codebook[1], continuous_weight[2] (unused: STE forward
// value == quantized), bias[3], indices[4].  ws: Wt (32MB) | Xbf (8MB).
// ---------------------------------------------------------------------------
extern "C" void kernel_launch(void* const* d_in, const int* in_sizes, int n_in,
                              void* d_out, int out_size, void* d_ws, size_t ws_size,
                              hipStream_t stream) {
    const float* x        = (const float*)d_in[0];
    const float* codebook = (const float*)d_in[1];
    const float* bias     = (const float*)d_in[3];
    const int*   indices  = (const int*)d_in[4];
    float*       out      = (float*)d_out;

    __bf16* wt = (__bf16*)d_ws;                                      // [4096][4096]
    __bf16* xb = (__bf16*)((char*)d_ws + (size_t)N_DIM * K_DIM * 2); // [1024][4096]

    crom_dequant<<<dim3(N_DIM / 256, K_DIM / 8), 256, 0, stream>>>(codebook, indices, wt);
    crom_xconv<<<dim3((M_DIM * K_DIM) / 2048), 256, 0, stream>>>(x, xb);
    crom_gemm<<<dim3(N_DIM / BN, M_DIM / BM), 256, 0, stream>>>(xb, wt, bias, out);
}
